// MultiheadAttention_58007828300517
// MI455X (gfx1250) — matrix operand order repaired and checked by reading, not compiled, not run on hardware
//
#include <hip/hip_runtime.h>
#include <cstdint>
#include <cstddef>

// ---------------------------------------------------------------------------
// MI455X (gfx1250) multi-head attention, bf16 WMMA pipeline.
//   stage 1: fp32 -> bf16 converts (q/k/v, Wq/Wk/Wv/Wo)
//   stage 2: projections  qproj/kproj/vproj = X @ W^T   (v_wmma_f32_16x16x32_bf16)
//   stage 3: flash-style attention with post-softmax mask renormalization
//            (cross-lane reductions via DPP16 butterflies, not ds_bpermute)
//   stage 4: output projection (fp32 result)
// ---------------------------------------------------------------------------

typedef __attribute__((ext_vector_type(16))) __bf16 v16bf;
typedef __attribute__((ext_vector_type(8)))  float  v8f;

#define DEV __device__ __forceinline__

// CDNA5 wave32 fragment layouts make A and B^T fragments two contiguous
// 16-byte chunks per lane (CDNA5 ISA 7.12.2):
//   A (16x32, m = lane&15): elems 0-7 -> k = half*8+0..7, elems 8-15 -> 16+half*8+0..7
//   B (32x16, n = lane&15): elems 0-7 -> k = half*8+0..7, elems 8-15 -> 16+half*8+0..7 (same as A)
DEV v16bf frag_pair(const __bf16* p0, const __bf16* p1) {
  union { v16bf v; uint4 q[2]; } u;
  u.q[0] = *(const uint4*)p0;
  u.q[1] = *(const uint4*)p1;
  return u.v;
}

DEV v8f wmma_bf16(v16bf a, v16bf b, v8f c) {
  // (neg_a, A, neg_b, B, c_mod, C, reuse_a, reuse_b)
  return __builtin_amdgcn_wmma_f32_16x16x32_bf16(false, a, false, b, (short)0, c,
                                                 false, false);
}

// ---- DPP16 butterfly reduction within a 16-lane half (DPP row) -------------
// quad_perm[1,0,3,2]=0xB1 (xor1), quad_perm[2,3,0,1]=0x4E (xor2),
// ROW_HALF_MIRROR=0x141 (xor across 4s), ROW_MIRROR=0x140 (xor across 8s).
template <int CTRL>
DEV float dppmov(float x) {
  return __int_as_float(__builtin_amdgcn_update_dpp(
      0, __float_as_int(x), CTRL, 0xF, 0xF, true));
}
DEV float redmax16(float x) {
  x = fmaxf(x, dppmov<0xB1>(x));
  x = fmaxf(x, dppmov<0x4E>(x));
  x = fmaxf(x, dppmov<0x141>(x));
  x = fmaxf(x, dppmov<0x140>(x));
  return x;
}
DEV float redsum16(float x) {
  x += dppmov<0xB1>(x);
  x += dppmov<0x4E>(x);
  x += dppmov<0x141>(x);
  x += dppmov<0x140>(x);
  return x;
}

// ---------------------------------------------------------------------------
__global__ void cvt_bf16_kernel(const float* __restrict__ in,
                                __bf16* __restrict__ out, int n) {
  int i = (blockIdx.x * blockDim.x + threadIdx.x) * 4;
  if (i + 3 < n) {
    float4 f = *(const float4*)(in + i);
    out[i + 0] = (__bf16)f.x;
    out[i + 1] = (__bf16)f.y;
    out[i + 2] = (__bf16)f.z;
    out[i + 3] = (__bf16)f.w;
  }
}

// ---------------------------------------------------------------------------
// C[m][n] = sum_k A[m][k] * Bt[n][k].  A: MxK row-major bf16, Bt: NxK row-major.
// Block: 256 threads (8 waves), tile 256(M) x 64(N), k-step 32.
// Each wave: 32 rows (2 A-fragments) x 64 cols (4 B-fragments) = 8 WMMA/k-step.
template <bool F32OUT>
__global__ __launch_bounds__(256) void gemm_bt_kernel(
    const __bf16* __restrict__ A, const __bf16* __restrict__ Bt,
    void* __restrict__ Cout, int M, int Nc, int K) {
  __shared__ alignas(16) __bf16 Asm[256 * 32];  // 16KB
  __shared__ alignas(16) __bf16 Bsm[64 * 32];   //  4KB

  const int tid  = threadIdx.x;
  const int lane = tid & 31;
  const int w    = tid >> 5;        // wave 0..7
  const int hlf  = lane >> 4;       // 0/1
  const int l16  = lane & 15;
  const int mtile = (int)blockIdx.x * 256;
  const int ntile = (int)blockIdx.y * 64;

  v8f acc[2][4];
  v8f zero = {};
#pragma unroll
  for (int r2 = 0; r2 < 2; ++r2)
#pragma unroll
    for (int e = 0; e < 4; ++e) acc[r2][e] = zero;

  for (int kb = 0; kb < K; kb += 32) {
    __syncthreads();
    // stage A tile 256x32 (16KB): 1024 chunks of 8 bf16, 4 per thread
#pragma unroll
    for (int j = 0; j < 4; ++j) {
      int c = tid + 256 * j;
      int r = c >> 2, off = (c & 3) * 8;
      *(uint4*)&Asm[r * 32 + off] =
          *(const uint4*)&A[(size_t)(mtile + r) * K + kb + off];
    }
    // stage Bt tile 64x32 (4KB): 256 chunks, 1 per thread
    {
      int r = tid >> 2, off = (tid & 3) * 8;
      *(uint4*)&Bsm[r * 32 + off] =
          *(const uint4*)&Bt[(size_t)(ntile + r) * K + kb + off];
    }
    __syncthreads();

    const __bf16* arow0 = &Asm[(w * 32 + l16) * 32];
    const __bf16* arow1 = arow0 + 16 * 32;
    v16bf af0 = frag_pair(arow0 + hlf * 8, arow0 + 16 + hlf * 8);
    v16bf af1 = frag_pair(arow1 + hlf * 8, arow1 + 16 + hlf * 8);
#pragma unroll
    for (int ef = 0; ef < 4; ++ef) {
      const __bf16* brow = &Bsm[(ef * 16 + l16) * 32];
      v16bf bfv = frag_pair(brow + hlf * 8, brow + 16 + hlf * 8);
      acc[0][ef] = wmma_bf16(af0, bfv, acc[0][ef]);
      acc[1][ef] = wmma_bf16(af1, bfv, acc[1][ef]);
    }
  }

  // C layout: element i -> row = i + 8*hlf, col = lane&15
#pragma unroll
  for (int r2 = 0; r2 < 2; ++r2) {
    const int mrow = mtile + w * 32 + r2 * 16 + hlf * 8;
#pragma unroll
    for (int ef = 0; ef < 4; ++ef) {
      int col = ntile + ef * 16 + l16;
#pragma unroll
      for (int i = 0; i < 8; ++i) {
        size_t idx = (size_t)(mrow + i) * Nc + col;
        if (F32OUT) ((float*)Cout)[idx] = acc[r2][ef][i];
        else        ((__bf16*)Cout)[idx] = (__bf16)acc[r2][ef][i];
      }
    }
  }
}

// ---------------------------------------------------------------------------
// Flash-style attention with post-softmax mask + renormalization:
//   att = e*m / (Sm + EPS*Z),  Z = sum e,  Sm = sum e*m  (running-max rescaled)
// Block: 128 threads (4 waves) per (h, n, 64-row q-tile). Each wave: 16 q rows.
__global__ __launch_bounds__(128) void attn_kernel(
    const __bf16* __restrict__ qp,   // [N*NQ][1024]  (col = h*64+e)
    const __bf16* __restrict__ kp,   // [N*NK][1024]
    const __bf16* __restrict__ vp,   // [N*NK][1024]
    const float*  __restrict__ mask, // [H*N][NQ][NK] fp32
    __bf16* __restrict__ oc)         // [N*NQ][1024]
{
  __shared__ alignas(16) __bf16 Vls[64 * 32];     // V^T chunk: [e][key]
  __shared__ alignas(16) __bf16 Pls[4][16 * 32];  // per-wave P tile

  const int tid  = threadIdx.x;
  const int lane = tid & 31;
  const int w    = tid >> 5;   // 0..3
  const int hlf  = lane >> 4;
  const int l16  = lane & 15;

  const int b  = blockIdx.x;
  const int qt = b & 15;          // NQ/64 = 16
  const int n  = (b >> 4) & 7;    // N = 8
  const int h  = b >> 7;          // H = 16

  const int qrow0 = n * 1024 + qt * 64 + w * 16;  // global row base for wave
  const int hoff  = h * 64;

  // Q fragments (persist in registers): A is 16x64 over two k-steps
  v16bf qa[2];
  {
    const __bf16* qr = qp + (size_t)(qrow0 + l16) * 1024 + hoff;
#pragma unroll
    for (int s = 0; s < 2; ++s)
      qa[s] = frag_pair(qr + s * 32 + hlf * 8, qr + s * 32 + 16 + hlf * 8);
  }

  float mrun[8], Zr[8], Smr[8];
#pragma unroll
  for (int i = 0; i < 8; ++i) { mrun[i] = -1e30f; Zr[i] = 0.f; Smr[i] = 0.f; }
  v8f of[4];
  v8f zero = {};
#pragma unroll
  for (int e = 0; e < 4; ++e) of[e] = zero;

  const float* mrow =
      mask + ((size_t)(h * 8 + n) * 1024 + (qt * 64 + w * 16)) * 1024;

  for (int kc = 0; kc < 1024; kc += 32) {
    // ---- cooperative stage of V^T chunk (32 keys x 64 e -> [e][key]) ----
    __syncthreads();
    {
      int keyl = tid >> 2;             // 0..31
      int e0   = (tid & 3) * 16;       // 0,16,32,48
      const __bf16* vr = vp + (size_t)(n * 1024 + kc + keyl) * 1024 + hoff + e0;
      uint4 d0 = *(const uint4*)(vr);
      uint4 d1 = *(const uint4*)(vr + 8);
      const __bf16* pv0 = (const __bf16*)&d0;
      const __bf16* pv1 = (const __bf16*)&d1;
#pragma unroll
      for (int j = 0; j < 8; ++j) Vls[(e0 + j) * 32 + keyl] = pv0[j];
#pragma unroll
      for (int j = 0; j < 8; ++j) Vls[(e0 + 8 + j) * 32 + keyl] = pv1[j];
    }
    __syncthreads();

    // ---- S = Q K^T : kproj rows are B^T rows (k = feature e) ----
    v8f sc[2];
#pragma unroll
    for (int t = 0; t < 2; ++t) {
      v8f c = zero;
#pragma unroll
      for (int s = 0; s < 2; ++s) {
        const __bf16* kr =
            kp + (size_t)(n * 1024 + kc + t * 16 + l16) * 1024 + hoff + s * 32;
        v16bf bkv = frag_pair(kr + hlf * 8, kr + 16 + hlf * 8);
        c = wmma_bf16(qa[s], bkv, c);
      }
      sc[t] = c;
    }

    // prefetch next mask chunk (global_prefetch_b8)
    if (kc + 32 < 1024)
      __builtin_prefetch(mrow + (size_t)(8 * hlf) * 1024 + kc + 32 + l16, 0, 0);

    // ---- online softmax + mask; row reductions are DPP16 butterflies -----
    float s0[8], s1[8], mnew[8], alpha[8], p0[8], p1[8];
#pragma unroll
    for (int i = 0; i < 8; ++i) { s0[i] = sc[0][i] * 0.125f; s1[i] = sc[1][i] * 0.125f; }
#pragma unroll
    for (int i = 0; i < 8; ++i) {
      float mx = redmax16(fmaxf(s0[i], s1[i]));
      mnew[i]  = fmaxf(mrun[i], mx);
      alpha[i] = __expf(mrun[i] - mnew[i]);
      mrun[i]  = mnew[i];
    }
#pragma unroll
    for (int i = 0; i < 8; ++i) {
      float e0 = __expf(s0[i] - mnew[i]);
      float e1 = __expf(s1[i] - mnew[i]);
      const float* mp = mrow + (size_t)(i + 8 * hlf) * 1024 + kc;
      float m0 = mp[l16], m1 = mp[16 + l16];
      p0[i] = e0 * m0;
      p1[i] = e1 * m1;
      float zs = redsum16(e0 + e1);
      float ss = redsum16(p0[i] + p1[i]);
      Zr[i]  = Zr[i]  * alpha[i] + zs;
      Smr[i] = Smr[i] * alpha[i] + ss;
    }
#pragma unroll
    for (int ef = 0; ef < 4; ++ef)
#pragma unroll
      for (int i = 0; i < 8; ++i) of[ef][i] *= alpha[i];

    // ---- relayout P (C layout -> A layout) through per-wave LDS tile ----
    __bf16* pb = &Pls[w][0];
#pragma unroll
    for (int i = 0; i < 8; ++i) {
      pb[(i + 8 * hlf) * 32 + l16]      = (__bf16)p0[i];
      pb[(i + 8 * hlf) * 32 + 16 + l16] = (__bf16)p1[i];
    }
    const __bf16* pr = pb + l16 * 32;
    v16bf pa = frag_pair(pr + hlf * 8, pr + 16 + hlf * 8);

    // ---- O += P @ V  (B^T rows from Vls) ----
#pragma unroll
    for (int ef = 0; ef < 4; ++ef) {
      const __bf16* vrw = &Vls[(ef * 16 + l16) * 32];
      v16bf bvv = frag_pair(vrw + hlf * 8, vrw + 16 + hlf * 8);
      of[ef] = wmma_bf16(pa, bvv, of[ef]);
    }
  }

  // ---- finalize: att = e*m / (Sm + EPS*Z) ----
  float inv[8];
#pragma unroll
  for (int i = 0; i < 8; ++i) inv[i] = 1.0f / (Smr[i] + 1e-6f * Zr[i]);
#pragma unroll
  for (int ef = 0; ef < 4; ++ef) {
    int col = hoff + ef * 16 + l16;
#pragma unroll
    for (int i = 0; i < 8; ++i) {
      int qrow = qrow0 + i + 8 * hlf;
      oc[(size_t)qrow * 1024 + col] = (__bf16)(of[ef][i] * inv[i]);
    }
  }
}

// ---------------------------------------------------------------------------
extern "C" void kernel_launch(void* const* d_in, const int* in_sizes, int n_in,
                              void* d_out, int out_size, void* d_ws, size_t ws_size,
                              hipStream_t stream) {
  (void)in_sizes; (void)n_in; (void)out_size; (void)ws_size;
  const float* q    = (const float*)d_in[0];
  const float* k    = (const float*)d_in[1];
  const float* v    = (const float*)d_in[2];
  const float* mask = (const float*)d_in[3];
  const float* Wq   = (const float*)d_in[4];
  const float* Wk   = (const float*)d_in[5];
  const float* Wv   = (const float*)d_in[6];
  const float* Wo   = (const float*)d_in[7];

  const int M  = 8 * 1024;          // N*NQ rows
  const int D  = 1024;
  const int nQKV = M * D;           // 8M elements
  const int nW   = D * D;           // 1M elements

  char* ws = (char*)d_ws;
  size_t off = 0;
  auto alloc = [&](size_t elems) {
    __bf16* p = (__bf16*)(ws + off);
    off += elems * sizeof(__bf16);
    return p;
  };
  __bf16* Qb  = alloc((size_t)nQKV);
  __bf16* Kb  = alloc((size_t)nQKV);
  __bf16* Vb  = alloc((size_t)nQKV);
  __bf16* Wqb = alloc((size_t)nW);   // [h*64+e][d]  == B^T for projection
  __bf16* Wkb = alloc((size_t)nW);
  __bf16* Wvb = alloc((size_t)nW);
  __bf16* Wob = alloc((size_t)nW);   // [j][i]       == B^T for out proj
  __bf16* qpj = alloc((size_t)nQKV);
  __bf16* kpj = alloc((size_t)nQKV);
  __bf16* vpj = alloc((size_t)nQKV);
  __bf16* oc  = alloc((size_t)nQKV);

  // stage 1: converts
  cvt_bf16_kernel<<<nQKV / (256 * 4), 256, 0, stream>>>(q, Qb, nQKV);
  cvt_bf16_kernel<<<nQKV / (256 * 4), 256, 0, stream>>>(k, Kb, nQKV);
  cvt_bf16_kernel<<<nQKV / (256 * 4), 256, 0, stream>>>(v, Vb, nQKV);
  cvt_bf16_kernel<<<nW   / (256 * 4), 256, 0, stream>>>(Wq, Wqb, nW);
  cvt_bf16_kernel<<<nW   / (256 * 4), 256, 0, stream>>>(Wk, Wkb, nW);
  cvt_bf16_kernel<<<nW   / (256 * 4), 256, 0, stream>>>(Wv, Wvb, nW);
  cvt_bf16_kernel<<<nW   / (256 * 4), 256, 0, stream>>>(Wo, Wob, nW);

  // stage 2: projections (M=8192, N=1024, K=1024)
  dim3 ggrid(M / 256, D / 64);
  gemm_bt_kernel<false><<<ggrid, 256, 0, stream>>>(Qb, Wqb, qpj, M, D, D);
  gemm_bt_kernel<false><<<ggrid, 256, 0, stream>>>(Kb, Wkb, kpj, M, D, D);
  gemm_bt_kernel<false><<<ggrid, 256, 0, stream>>>(Vb, Wvb, vpj, M, D, D);

  // stage 3: attention  (H*N*(NQ/64) = 16*8*16 = 2048 blocks)
  attn_kernel<<<2048, 128, 0, stream>>>(qpj, kpj, vpj, mask, oc);

  // stage 4: output projection -> fp32 d_out
  gemm_bt_kernel<true><<<ggrid, 256, 0, stream>>>(oc, Wob, d_out, M, D, D);
}
